// LSTM_80908593923379
// MI455X (gfx1250) — matrix operand order repaired and checked by reading, not compiled, not run on hardware
//
#include <hip/hip_runtime.h>
#include <hip/hip_bf16.h>

constexpr int kH   = 1024;  // HIDDEN
constexpr int kB   = 512;   // BATCH
constexpr int kSeq = 256;   // SEQ_LEN
constexpr int kNC  = 10;    // NUM_CLASSES
// LDS layout written by TDM: 8 bytes of pad after every 1024 bytes of payload.
// Row (1024 bf16 = 2048B) => row stride 2064B = 1032 elements; elements with
// k >= 512 are shifted by +4 elements (8B) inside the row.
constexpr int kPK  = 1032;

typedef __attribute__((ext_vector_type(16))) __bf16 v16bf;
typedef __attribute__((ext_vector_type(8)))  __bf16 v8bf;
typedef __attribute__((ext_vector_type(8)))  float  v8f;
typedef __attribute__((ext_vector_type(4)))  unsigned int u32x4;
typedef __attribute__((ext_vector_type(8)))  int          i32x8;
typedef __attribute__((ext_vector_type(4)))  int          i32x4;

union Frag16 { v16bf v; v8bf h[2]; };

__device__ __forceinline__ unsigned short f2bf_u16(float f) {
  unsigned int u = __float_as_uint(f);
  u += 0x7FFFu + ((u >> 16) & 1u);       // round-to-nearest-even
  return (unsigned short)(u >> 16);
}

__device__ __forceinline__ float bf2f(unsigned short s) {
  return __uint_as_float(((unsigned int)s) << 16);
}

// ---- convert 4 recurrent weight matrices f32 -> bf16, packed [4][H][H] ----
__global__ void cvt_weights(const float* __restrict__ Wg,
                            const float* __restrict__ Wi,
                            const float* __restrict__ Wf,
                            const float* __restrict__ Wo,
                            unsigned short* __restrict__ out) {
  int idx = blockIdx.x * blockDim.x + threadIdx.x;  // 4*H*H threads exactly
  int gate = idx >> 20;                             // H*H = 2^20
  int rem  = idx & (kH * kH - 1);
  const float* src = (gate == 0) ? Wg : (gate == 1) ? Wi : (gate == 2) ? Wf : Wo;
  out[idx] = f2bf_u16(src[rem]);
}

// ---- zero h0 (bf16 transposed) and c0 (f32) ----
__global__ void init_state(unsigned short* __restrict__ hT0,
                           float* __restrict__ c) {
  int idx = blockIdx.x * blockDim.x + threadIdx.x;  // H*B threads exactly
  hT0[idx] = 0;
  c[idx]   = 0.0f;
}

// ---- one LSTM timestep ----
// grid = (H/16, B/(16*8)) = (64, 4), block = 256 threads (8 waves).
// All 8 waves share the same 16 hidden rows -> DMA the 4-gate weight slab
// (4 x 16 x 1024 bf16 = 128 KB + pad) into WGP LDS with ONE tensor_load_to_lds
// (TDM, TENSORcnt) per block; WMMA A-operands then come from LDS.
// Each wave: one 16(hidden) x 16(batch) tile for ALL four gates.
__global__ __launch_bounds__(256)
void lstm_step(const unsigned short* __restrict__ Wbf,   // [4][H][H] bf16 (g,i,f,o)
               const unsigned short* __restrict__ hTin,  // [B][H] bf16 (h transposed)
               unsigned short* __restrict__ hTout,       // [B][H] bf16
               float* __restrict__ cst,                  // [H][B] f32
               const float* __restrict__ x,              // [B][SEQ] f32
               const float* __restrict__ wgx, const float* __restrict__ bg,
               const float* __restrict__ wix, const float* __restrict__ bi,
               const float* __restrict__ wfx, const float* __restrict__ bfv,
               const float* __restrict__ wox, const float* __restrict__ bo,
               int t) {
  extern __shared__ unsigned short sW[];               // [4*16][kPK] = 129 KiB

  const int tid  = threadIdx.x;
  const int lane = tid & 31;
  const int wave = tid >> 5;
  const int m0   = blockIdx.x * 16;                    // hidden tile (shared by block)
  const int n0   = (blockIdx.y * 8 + wave) * 16;       // batch tile (per wave)
  const int sub  = lane & 15;
  const int half = lane >> 4;

  const __bf16* hT = (const __bf16*)hTin;

  // ---- TDM stage: one descriptor DMAs the whole 3D slab Global->LDS ----
  if (wave == 0) {
    const unsigned long long ga =
        (unsigned long long)(uintptr_t)Wbf + (unsigned long long)m0 * kH * 2ull;
    const unsigned int lds0 = (unsigned int)(uintptr_t)(&sW[0]);  // low 32b = LDS offset

    u32x4 g0;
    g0.x = 1u;                                     // count=1 (valid), no gather
    g0.y = lds0;                                   // lds_addr (bytes)
    g0.z = (unsigned int)(ga & 0xFFFFFFFFull);     // global_addr[31:0]
    g0.w = (unsigned int)((ga >> 32) & 0x1FFFFFFull) | (2u << 30);  // addr[56:32] | type=2

    i32x8 g1;
    g1[0] = (1 << 16)      // data_size = 1 (2 bytes)
          | (1 << 20)      // pad_enable
          | (7 << 22)      // pad_interval: 256 DWORDs = 1024B
          | (1 << 25);     // pad_amount: 2 DWORDs = 8B
    g1[1] = (1024 << 16);  // tensor_dim0[15:0]=1024 (atomic_barrier_addr=0)
    g1[2] = (16 << 16);    // tensor_dim0[31:16]=0, tensor_dim1[15:0]=16
    g1[3] = (1024 << 16);  // tensor_dim1[31:16]=0, tile_dim0=1024
    g1[4] = 16 | (4 << 16);// tile_dim1=16, tile_dim2=4 (gates)
    g1[5] = 1024;          // tensor_dim0_stride[31:0] (row stride, elements)
    g1[6] = 0;             // dim0_stride[47:32]=0, tensor_dim1_stride[15:0]=0 (H*H low16)
    g1[7] = 16;            // tensor_dim1_stride[47:16] = (H*H)>>16 = 16

    i32x4 g2;
    g2[0] = 4;             // tensor_dim2 = 4 gates
    g2[1] = 0;             // tensor_dim3 (unused)
    g2[2] = kH * kH;       // tensor_dim2_stride[31:0] = gate stride (elements)
    g2[3] = 0;             // dim2_stride[47:32]=0, tile_dim3=0

    i32x4 g3 = {0, 0, 0, 0};
    i32x8 g4 = {0, 0, 0, 0, 0, 0, 0, 0};           // VADDR4: unused / NULL

    __builtin_amdgcn_tensor_load_to_lds(g0, g1, g2, g3, g4, 0);
    __builtin_amdgcn_s_wait_tensorcnt(0);
  }
  __syncthreads();

  // A layout (16-bit, 16x32): lanes 0-15 hold M=sub, K in {0..7, 16..23};
  // lanes 16-31 hold K in {8..15, 24..31}  -> per-lane base offset half*8, hi +16.
  // B layout (32x16): same K order as A: lanes 0-15 K in {0..7, 16..23}, lanes 16-31 K in {8..15, 24..31}.
  const size_t brow = (size_t)(n0 + sub) * kH;
  const int aG_base = (0 * 16 + sub) * kPK;
  const int aI_base = (1 * 16 + sub) * kPK;
  const int aF_base = (2 * 16 + sub) * kPK;
  const int aO_base = (3 * 16 + sub) * kPK;

  v8f aG = {}, aI = {}, aF = {}, aO = {};

  for (int k0 = 0; k0 < kH; k0 += 32) {
    // prefetch the B (h) stream a few iterations ahead -> global_prefetch_b8
    if (k0 + 128 < kH) __builtin_prefetch((const void*)(hT + brow + k0 + 128), 0, 1);

    Frag16 bF;
    const __bf16* bp = hT + brow + k0 + half * 8;
    bF.h[0] = *(const v8bf*)(bp);
    bF.h[1] = *(const v8bf*)(bp + 16);

    // +4-element shift for the second half of each padded LDS row
    const int ka = k0 + half * 8 + ((k0 >= 512) ? 4 : 0);
    Frag16 ag, ai, af, ao;
    ag.h[0] = *(const v8bf*)(&sW[aG_base + ka]); ag.h[1] = *(const v8bf*)(&sW[aG_base + ka + 16]);
    ai.h[0] = *(const v8bf*)(&sW[aI_base + ka]); ai.h[1] = *(const v8bf*)(&sW[aI_base + ka + 16]);
    af.h[0] = *(const v8bf*)(&sW[aF_base + ka]); af.h[1] = *(const v8bf*)(&sW[aF_base + ka + 16]);
    ao.h[0] = *(const v8bf*)(&sW[aO_base + ka]); ao.h[1] = *(const v8bf*)(&sW[aO_base + ka + 16]);

    aG = __builtin_amdgcn_wmma_f32_16x16x32_bf16(false, ag.v, false, bF.v, (short)0, aG, false, false);
    aI = __builtin_amdgcn_wmma_f32_16x16x32_bf16(false, ai.v, false, bF.v, (short)0, aI, false, false);
    aF = __builtin_amdgcn_wmma_f32_16x16x32_bf16(false, af.v, false, bF.v, (short)0, aF, false, false);
    aO = __builtin_amdgcn_wmma_f32_16x16x32_bf16(false, ao.v, false, bF.v, (short)0, aO, false, false);
  }

  // D layout: VGPR r, lane l -> m = m0 + r + 8*(l>>4), n = n0 + (l&15)
  const int n = n0 + sub;
  const float xv = x[(size_t)n * kSeq + t];

#pragma unroll
  for (int r = 0; r < 8; ++r) {
    const int m = m0 + r + half * 8;
    float gp = aG[r] + wgx[m] * xv + bg[m];
    float ip = aI[r] + wix[m] * xv + bi[m];
    float fp = aF[r] + wfx[m] * xv + bfv[m];
    float op = aO[r] + wox[m] * xv + bo[m];

    float g = tanhf(gp);
    float i = 1.0f / (1.0f + expf(-ip));
    float f = 1.0f / (1.0f + expf(-fp));
    float o = 1.0f / (1.0f + expf(-op));

    const size_t cidx = (size_t)m * kB + n;
    float cn = g * i + cst[cidx] * f;   // in-place: (m,n) owned by this wave only
    cst[cidx] = cn;
    float hn = tanhf(cn) * o;
    hTout[(size_t)n * kH + m] = f2bf_u16(hn);
  }
}

// ---- final projection: out[b][c] = Wph[c,:] . h[:,b] + bp[c] ----
__global__ void proj_kernel(const unsigned short* __restrict__ hT,  // [B][H] bf16
                            const float* __restrict__ Wph,          // [NC][H]
                            const float* __restrict__ bp,
                            float* __restrict__ out) {              // [B][NC]
  int idx = blockIdx.x * blockDim.x + threadIdx.x;
  if (idx >= kB * kNC) return;
  int b = idx / kNC;
  int c = idx % kNC;
  const unsigned short* h = hT + (size_t)b * kH;
  const float* w = Wph + (size_t)c * kH;
  float s = 0.0f;
  for (int k = 0; k < kH; ++k) s += w[k] * bf2f(h[k]);
  out[idx] = s + bp[c];
}

extern "C" void kernel_launch(void* const* d_in, const int* in_sizes, int n_in,
                              void* d_out, int out_size, void* d_ws, size_t ws_size,
                              hipStream_t stream) {
  (void)in_sizes; (void)n_in; (void)out_size; (void)ws_size;

  const float* x   = (const float*)d_in[0];
  const float* Wgx = (const float*)d_in[1];
  const float* Wgh = (const float*)d_in[2];
  const float* bg  = (const float*)d_in[3];
  const float* Wix = (const float*)d_in[4];
  const float* Wih = (const float*)d_in[5];
  const float* bi  = (const float*)d_in[6];
  const float* Wfx = (const float*)d_in[7];
  const float* Wfh = (const float*)d_in[8];
  const float* bfv = (const float*)d_in[9];
  const float* Wox = (const float*)d_in[10];
  const float* Woh = (const float*)d_in[11];
  const float* bo  = (const float*)d_in[12];
  const float* Wph = (const float*)d_in[13];
  const float* bp  = (const float*)d_in[14];

  // workspace layout: [4*H*H bf16 weights][hT0][hT1][c f32]  (~12.6 MB)
  unsigned short* wbf = (unsigned short*)d_ws;
  unsigned short* hT0 = wbf + (size_t)4 * kH * kH;
  unsigned short* hT1 = hT0 + (size_t)kB * kH;
  float*          cst = (float*)(hT1 + (size_t)kB * kH);

  cvt_weights<<<(4 * kH * kH) / 256, 256, 0, stream>>>(Wgh, Wih, Wfh, Woh, wbf);
  init_state<<<(kH * kB) / 256, 256, 0, stream>>>(hT0, cst);

  const size_t lds_bytes = (size_t)4 * 16 * kPK * sizeof(unsigned short);  // 132096
  dim3 grid(kH / 16, kB / (16 * 8));  // (64, 4) -> 256 blocks, 8 waves each
  for (int t = 0; t < kSeq; ++t) {
    const unsigned short* hin = (t & 1) ? hT1 : hT0;
    unsigned short*      hout = (t & 1) ? hT0 : hT1;
    lstm_step<<<grid, 256, lds_bytes, stream>>>(wbf, hin, hout, cst, x,
                                                Wgx, bg, Wix, bi, Wfx, bfv, Wox, bo, t);
  }
  // SEQ even: final h is in hT0
  proj_kernel<<<(kB * kNC + 255) / 256, 256, 0, stream>>>(hT0, Wph, bp, (float*)d_out);
}
